// GraphElementNetwork_26268019982724
// MI455X (gfx1250) — matrix operand order repaired and checked by reading, not compiled, not run on hardware
//
#include <hip/hip_runtime.h>

typedef __attribute__((ext_vector_type(16))) _Float16 v16h;
typedef __attribute__((ext_vector_type(8)))  _Float16 v8h;
typedef __attribute__((ext_vector_type(8)))  float    v8f;

#define TPB 256
#define WPB 8   // waves per block (wave32)

__device__ __forceinline__ v8f wmma_f32_16x16x32(v16h a, v16h b, v8f c) {
  // (neg_a, A, neg_b, B, c_mod, C, reuse_a, reuse_b)
  return __builtin_amdgcn_wmma_f32_16x16x32_f16(false, a, false, b, (short)0, c,
                                                false, false);
}

// load 16 contiguous f16 from LDS (16B-aligned) into a v16h fragment
__device__ __forceinline__ v16h ld16h(const _Float16* p) {
  v8h lo = *(const v8h*)p;
  v8h hi = *(const v8h*)(p + 8);
  v16h r;
#pragma unroll
  for (int i = 0; i < 8; ++i) { r[i] = lo[i]; r[i + 8] = hi[i]; }
  return r;
}

// ----------------------------- utility kernels -----------------------------

__global__ void k_fill_u32(unsigned* __restrict__ p, unsigned v, int n) {
  int i = blockIdx.x * blockDim.x + threadIdx.x;
  if (i < n) p[i] = v;
}

// order-preserving float->uint map so atomicMax(uint) == float max
__global__ void k_edge_max(const float* __restrict__ dist, const int* __restrict__ dst,
                           unsigned* __restrict__ m, int E) {
  int e = blockIdx.x * blockDim.x + threadIdx.x;
  if (e >= E) return;
  unsigned u = __float_as_uint(dist[e]);
  u = (u & 0x80000000u) ? ~u : (u | 0x80000000u);
  atomicMax(&m[dst[e]], u);
}

__global__ void k_exp_sum(const float* __restrict__ dist, const int* __restrict__ dst,
                          const unsigned* __restrict__ m, float* __restrict__ wbuf,
                          float* __restrict__ sbuf, int E) {
  int e = blockIdx.x * blockDim.x + threadIdx.x;
  if (e >= E) return;
  int d = dst[e];
  unsigned u = m[d];
  u = (u & 0x80000000u) ? (u & 0x7fffffffu) : ~u;
  float ev = expf(dist[e] - __uint_as_float(u));
  wbuf[e] = ev;
  atomicAdd(&sbuf[d], ev);
}

__global__ void k_norm(float* __restrict__ wbuf, const float* __restrict__ sbuf,
                       const int* __restrict__ dst, int E) {
  int e = blockIdx.x * blockDim.x + threadIdx.x;
  if (e >= E) return;
  wbuf[e] = wbuf[e] / sbuf[dst[e]];
}

// agg[dst] += w * h[src]  (h is [N,8] f32)
__global__ void k_wagg(const float* __restrict__ w, const int* __restrict__ src,
                       const int* __restrict__ dst, const float* __restrict__ h,
                       float* __restrict__ agg, int E) {
  int e = blockIdx.x * blockDim.x + threadIdx.x;
  if (e >= E) return;
  float wv = w[e];
  int s = src[e], d = dst[e];
  const float4* hp = (const float4*)(h + (size_t)s * 8);
  float4 a = hp[0], b = hp[1];
  float* o = agg + (size_t)d * 8;
  atomicAdd(o + 0, wv * a.x); atomicAdd(o + 1, wv * a.y);
  atomicAdd(o + 2, wv * a.z); atomicAdd(o + 3, wv * a.w);
  atomicAdd(o + 4, wv * b.x); atomicAdd(o + 5, wv * b.y);
  atomicAdd(o + 6, wv * b.z); atomicAdd(o + 7, wv * b.w);
}

// ---------------- scalar per-node 3-layer MLP (small FLOP share) ------------
// y[n] = relu(relu(relu([x1|x2] @ W1 + b1) @ W2 + b2) @ W3 + b3)
template <int DIN1, int DIN2, int DOUT>
__global__ void __launch_bounds__(TPB) k_mlp3(
    const float* __restrict__ x1, const float* __restrict__ x2,
    const float* __restrict__ W1, const float* __restrict__ B1,
    const float* __restrict__ W2, const float* __restrict__ B2,
    const float* __restrict__ W3, const float* __restrict__ B3,
    float* __restrict__ y, int N) {
  constexpr int DIN = DIN1 + DIN2;
  __shared__ float w1s[DIN * 64];
  __shared__ float w2s[64 * 64];
  __shared__ float w3s[64 * DOUT];
  __shared__ float b1s[64], b2s[64], b3s[DOUT];
  for (int i = threadIdx.x; i < DIN * 64; i += TPB) w1s[i] = W1[i];
  for (int i = threadIdx.x; i < 64 * 64; i += TPB) w2s[i] = W2[i];
  for (int i = threadIdx.x; i < 64 * DOUT; i += TPB) w3s[i] = W3[i];
  if (threadIdx.x < 64) { b1s[threadIdx.x] = B1[threadIdx.x]; b2s[threadIdx.x] = B2[threadIdx.x]; }
  if (threadIdx.x < DOUT) b3s[threadIdx.x] = B3[threadIdx.x];
  __syncthreads();

  int n = blockIdx.x * blockDim.x + threadIdx.x;
  if (n >= N) return;

  float xin[DIN];
#pragma unroll
  for (int i = 0; i < DIN1; ++i) xin[i] = x1[(size_t)n * DIN1 + i];
#pragma unroll
  for (int i = 0; i < DIN2; ++i) xin[DIN1 + i] = x2[(size_t)n * DIN2 + i];

  float h1[64];
#pragma unroll
  for (int j = 0; j < 64; ++j) {
    float a = b1s[j];
#pragma unroll
    for (int i = 0; i < DIN; ++i) a = fmaf(xin[i], w1s[i * 64 + j], a);
    h1[j] = fmaxf(a, 0.f);
  }

  float out[DOUT];
#pragma unroll
  for (int o = 0; o < DOUT; ++o) out[o] = b3s[o];
  for (int j = 0; j < 64; ++j) {   // fuse layer2 + layer3: no h2 array
    float acc = b2s[j];
#pragma unroll
    for (int i = 0; i < 64; ++i) acc = fmaf(h1[i], w2s[i * 64 + j], acc);
    acc = fmaxf(acc, 0.f);
#pragma unroll
    for (int o = 0; o < DOUT; ++o) out[o] = fmaf(acc, w3s[j * DOUT + o], out[o]);
  }
#pragma unroll
  for (int o = 0; o < DOUT; ++o) y[(size_t)n * DOUT + o] = fmaxf(out[o], 0.f);
}

// -------------- WMMA edge MLP: [he, h[src], h[dst]] (17) -> 64 -> 64 -> 1 ---
// One wave32 handles a tile of 16 edges; weights live in VGPRs as B fragments.
__global__ void __launch_bounds__(TPB) k_edge_mlp(
    const float* __restrict__ he_in, const float* __restrict__ h,
    const int* __restrict__ src, const int* __restrict__ dst,
    const float* __restrict__ W1, const float* __restrict__ B1,   // [17,64],[64]
    const float* __restrict__ W2, const float* __restrict__ B2,   // [64,64],[64]
    const float* __restrict__ W3, const float* __restrict__ B3,   // [64,1],[1]
    float* __restrict__ he_out, float* __restrict__ aggE, int E) {
  __shared__ __attribute__((aligned(32))) _Float16 w1t[64][32];   // W1^T, K padded 17->32
  __shared__ __attribute__((aligned(32))) _Float16 w2t[64][64];   // W2^T
  __shared__ __attribute__((aligned(32))) _Float16 w3t[16][64];   // W3 col 0, cols 1..15 zero
  __shared__ float b1s[64], b2s[64];
  __shared__ float b3s;
  __shared__ __attribute__((aligned(32))) _Float16 stage[WPB][16][64]; // per-wave staging

  int tid = threadIdx.x;
  for (int i = tid; i < 64 * 32; i += TPB) {
    int n = i >> 5, k = i & 31;
    w1t[n][k] = (k < 17) ? (_Float16)W1[k * 64 + n] : (_Float16)0.f;
  }
  for (int i = tid; i < 64 * 64; i += TPB) {
    int n = i >> 6, k = i & 63;
    w2t[n][k] = (_Float16)W2[k * 64 + n];
  }
  for (int i = tid; i < 16 * 64; i += TPB) {
    int n = i >> 6, k = i & 63;
    w3t[n][k] = (n == 0) ? (_Float16)W3[k] : (_Float16)0.f;
  }
  if (tid < 64) { b1s[tid] = B1[tid]; b2s[tid] = B2[tid]; }
  if (tid == 0) b3s = B3[0];
  __syncthreads();

  const int lane  = tid & 31;
  const int wave  = tid >> 5;
  const int col   = lane & 15;        // N index inside a 16-wide tile (B/C layout)
  const int khalf = (lane >> 4) & 1;  // which K half this lane holds
  const int m     = lane & 15;        // A-matrix row for this lane
  const int kb    = khalf * 16;       // A-fragment K base within a 32-K chunk

  // Preload all B fragments into VGPRs (held across the tile loop).
  v16h b1f[4], b2f[2][4], b3f[2];
#pragma unroll
  for (int nt = 0; nt < 4; ++nt)
    b1f[nt] = ld16h(&w1t[nt * 16 + col][khalf * 16]);
#pragma unroll
  for (int kc = 0; kc < 2; ++kc)
#pragma unroll
    for (int nt = 0; nt < 4; ++nt)
      b2f[kc][nt] = ld16h(&w2t[nt * 16 + col][kc * 32 + khalf * 16]);
#pragma unroll
  for (int kc = 0; kc < 2; ++kc)
    b3f[kc] = ld16h(&w3t[col][kc * 32 + khalf * 16]);
  const float b3v = b3s;

  _Float16 (*st)[64] = stage[wave];
  const int ntiles = (E + 15) >> 4;
  const int gwave  = blockIdx.x * WPB + wave;
  const int nwaves = gridDim.x * WPB;

  for (int tile = gwave; tile < ntiles; tile += nwaves) {
    const int ebase = tile << 4;

    // ---- gather: lanes 0..15 build one edge row each (f16, K padded to 32)
    if (lane < 16) {
      int e = ebase + lane;
      _Float16* row = st[lane];
      if (e < E) {
        float hev = he_in[e];
        int sn = src[e], dn = dst[e];
        const float4* hp = (const float4*)(h + (size_t)sn * 8);
        float4 s0 = hp[0], s1 = hp[1];
        hp = (const float4*)(h + (size_t)dn * 8);
        float4 d0 = hp[0], d1 = hp[1];
        row[0]  = (_Float16)hev;
        row[1]  = (_Float16)s0.x; row[2]  = (_Float16)s0.y;
        row[3]  = (_Float16)s0.z; row[4]  = (_Float16)s0.w;
        row[5]  = (_Float16)s1.x; row[6]  = (_Float16)s1.y;
        row[7]  = (_Float16)s1.z; row[8]  = (_Float16)s1.w;
        row[9]  = (_Float16)d0.x; row[10] = (_Float16)d0.y;
        row[11] = (_Float16)d0.z; row[12] = (_Float16)d0.w;
        row[13] = (_Float16)d1.x; row[14] = (_Float16)d1.y;
        row[15] = (_Float16)d1.z; row[16] = (_Float16)d1.w;
      } else {
#pragma unroll
        for (int k = 0; k < 17; ++k) row[k] = (_Float16)0.f;
      }
#pragma unroll
      for (int k = 17; k < 32; ++k) row[k] = (_Float16)0.f;
    }
    asm volatile("s_wait_dscnt 0" ::: "memory");

    // ---- layer 1: A[16x32] x W1 -> h1[16x64]
    v16h a1;
    {
      v8h lo = *(const v8h*)&st[m][kb];
      v8h hi = *(const v8h*)&st[m][kb + 8];
#pragma unroll
      for (int i = 0; i < 8; ++i) { a1[i] = lo[i]; a1[i + 8] = hi[i]; }
    }
    v8f c1[4];
#pragma unroll
    for (int nt = 0; nt < 4; ++nt) {
      float bv = b1s[nt * 16 + col];
      v8f c = {bv, bv, bv, bv, bv, bv, bv, bv};
      c = wmma_f32_16x16x32(a1, b1f[nt], c);
#pragma unroll
      for (int r = 0; r < 8; ++r) c[r] = fmaxf(c[r], 0.f);
      c1[nt] = c;
    }
    // restage h1 as f16 (C layout -> row-major 16x64)
#pragma unroll
    for (int nt = 0; nt < 4; ++nt)
#pragma unroll
      for (int r = 0; r < 8; ++r)
        st[khalf * 8 + r][nt * 16 + col] = (_Float16)c1[nt][r];
    asm volatile("s_wait_dscnt 0" ::: "memory");

    // ---- layer 2: A[16x64] x W2 -> h2[16x64]
    v16h a2[2];
#pragma unroll
    for (int kc = 0; kc < 2; ++kc) {
      v8h lo = *(const v8h*)&st[m][kc * 32 + kb];
      v8h hi = *(const v8h*)&st[m][kc * 32 + kb + 8];
#pragma unroll
      for (int i = 0; i < 8; ++i) { a2[kc][i] = lo[i]; a2[kc][i + 8] = hi[i]; }
    }
    v8f c2[4];
#pragma unroll
    for (int nt = 0; nt < 4; ++nt) {
      float bv = b2s[nt * 16 + col];
      v8f c = {bv, bv, bv, bv, bv, bv, bv, bv};
      c = wmma_f32_16x16x32(a2[0], b2f[0][nt], c);
      c = wmma_f32_16x16x32(a2[1], b2f[1][nt], c);
#pragma unroll
      for (int r = 0; r < 8; ++r) c[r] = fmaxf(c[r], 0.f);
      c2[nt] = c;
    }
#pragma unroll
    for (int nt = 0; nt < 4; ++nt)
#pragma unroll
      for (int r = 0; r < 8; ++r)
        st[khalf * 8 + r][nt * 16 + col] = (_Float16)c2[nt][r];
    asm volatile("s_wait_dscnt 0" ::: "memory");

    // ---- layer 3: A[16x64] x w3 -> [16x1]
    v16h a3[2];
#pragma unroll
    for (int kc = 0; kc < 2; ++kc) {
      v8h lo = *(const v8h*)&st[m][kc * 32 + kb];
      v8h hi = *(const v8h*)&st[m][kc * 32 + kb + 8];
#pragma unroll
      for (int i = 0; i < 8; ++i) { a3[kc][i] = lo[i]; a3[kc][i + 8] = hi[i]; }
    }
    v8f c3 = {b3v, b3v, b3v, b3v, b3v, b3v, b3v, b3v};
    c3 = wmma_f32_16x16x32(a3[0], b3f[0], c3);
    c3 = wmma_f32_16x16x32(a3[1], b3f[1], c3);

    // only column N==0 is meaningful: lanes 0 (rows 0..7) and 16 (rows 8..15)
    if (col == 0) {
      int mb = khalf * 8;
#pragma unroll
      for (int r = 0; r < 8; ++r) {
        int e = ebase + mb + r;
        if (e < E) {
          float v = fmaxf(c3[r], 0.f);
          he_out[e] = v;
          atomicAdd(&aggE[dst[e]], v);
        }
      }
    }
  }
}

// ------------------------------- launcher -----------------------------------
extern "C" void kernel_launch(void* const* d_in, const int* in_sizes, int n_in,
                              void* d_out, int out_size, void* d_ws, size_t ws_size,
                              hipStream_t stream) {
  (void)n_in; (void)out_size; (void)ws_size;
  const float* node_feat = (const float*)d_in[0];
  const float* edge_feat = (const float*)d_in[1];
  const float* edge_dist = (const float*)d_in[2];
  const int*   src       = (const int*)d_in[3];
  const int*   dst       = (const int*)d_in[4];
  const float* w_enc1 = (const float*)d_in[5];  const float* b_enc1 = (const float*)d_in[6];
  const float* w_enc2 = (const float*)d_in[7];  const float* b_enc2 = (const float*)d_in[8];
  const float* w_enc3 = (const float*)d_in[9];  const float* b_enc3 = (const float*)d_in[10];
  const float* w_dec1 = (const float*)d_in[11]; const float* b_dec1 = (const float*)d_in[12];
  const float* w_dec2 = (const float*)d_in[13]; const float* b_dec2 = (const float*)d_in[14];
  const float* w_dec3 = (const float*)d_in[15]; const float* b_dec3 = (const float*)d_in[16];
  const float* w_nod1 = (const float*)d_in[17]; const float* b_nod1 = (const float*)d_in[18];
  const float* w_nod2 = (const float*)d_in[19]; const float* b_nod2 = (const float*)d_in[20];
  const float* w_nod3 = (const float*)d_in[21]; const float* b_nod3 = (const float*)d_in[22];
  const float* w_edg1 = (const float*)d_in[23]; const float* b_edg1 = (const float*)d_in[24];
  const float* w_edg2 = (const float*)d_in[25]; const float* b_edg2 = (const float*)d_in[26];
  const float* w_edg3 = (const float*)d_in[27]; const float* b_edg3 = (const float*)d_in[28];

  const int N = in_sizes[0] / 3;
  const int E = in_sizes[3];

  char* ws = (char*)d_ws;
  size_t off = 0;
  auto alloc = [&](size_t bytes) -> void* {
    void* p = ws + off;
    off += (bytes + 255) & ~(size_t)255;
    return p;
  };
  float*    hA   = (float*)alloc((size_t)N * 8 * sizeof(float));
  float*    hB   = (float*)alloc((size_t)N * 8 * sizeof(float));
  float*    wbuf = (float*)alloc((size_t)E * sizeof(float));
  float*    he   = (float*)alloc((size_t)E * sizeof(float));
  float*    sbuf = (float*)alloc((size_t)N * sizeof(float));
  float*    aggE = (float*)alloc((size_t)N * sizeof(float));
  unsigned* mbuf = (unsigned*)alloc((size_t)N * sizeof(unsigned));

  const int gE  = (E + TPB - 1) / TPB;
  const int gN  = (N + TPB - 1) / TPB;
  const int gN8 = (N * 8 + TPB - 1) / TPB;
  int ntiles = (E + 15) / 16;
  int gW = (ntiles + WPB - 1) / WPB;
  if (gW > 2048) gW = 2048;
  if (gW < 1) gW = 1;

  // init segment-max (mapped -inf == 0) and segment-sum
  k_fill_u32<<<gN, TPB, 0, stream>>>(mbuf, 0u, N);
  k_fill_u32<<<gN, TPB, 0, stream>>>((unsigned*)sbuf, 0u, N);

  // encoder: node_feat [N,3] -> hB [N,8]
  k_mlp3<3, 0, 8><<<gN, TPB, 0, stream>>>(node_feat, nullptr,
      w_enc1, b_enc1, w_enc2, b_enc2, w_enc3, b_enc3, hB, N);

  // edge softmax weights over dst
  k_edge_max<<<gE, TPB, 0, stream>>>(edge_dist, dst, mbuf, E);
  k_exp_sum<<<gE, TPB, 0, stream>>>(edge_dist, dst, mbuf, wbuf, sbuf, E);
  k_norm<<<gE, TPB, 0, stream>>>(wbuf, sbuf, dst, E);

  // h = segment_sum(w * h_enc[src], dst) -> hA
  k_fill_u32<<<gN8, TPB, 0, stream>>>((unsigned*)hA, 0u, N * 8);
  k_wagg<<<gE, TPB, 0, stream>>>(wbuf, src, dst, hB, hA, E);

  // round 0
  k_fill_u32<<<gN, TPB, 0, stream>>>((unsigned*)aggE, 0u, N);
  k_edge_mlp<<<gW, TPB, 0, stream>>>(edge_feat, hA, src, dst,
      w_edg1, b_edg1, w_edg2, b_edg2, w_edg3, b_edg3, he, aggE, E);
  k_mlp3<1, 8, 8><<<gN, TPB, 0, stream>>>(aggE, hA,
      w_nod1, b_nod1, w_nod2, b_nod2, w_nod3, b_nod3, hB, N);

  // round 1
  k_fill_u32<<<gN, TPB, 0, stream>>>((unsigned*)aggE, 0u, N);
  k_edge_mlp<<<gW, TPB, 0, stream>>>(he, hB, src, dst,
      w_edg1, b_edg1, w_edg2, b_edg2, w_edg3, b_edg3, he, aggE, E);
  k_mlp3<1, 8, 8><<<gN, TPB, 0, stream>>>(aggE, hB,
      w_nod1, b_nod1, w_nod2, b_nod2, w_nod3, b_nod3, hA, N);

  // second softmax aggregation -> hB
  k_fill_u32<<<gN8, TPB, 0, stream>>>((unsigned*)hB, 0u, N * 8);
  k_wagg<<<gE, TPB, 0, stream>>>(wbuf, src, dst, hA, hB, E);

  // decoder: hB [N,8] -> d_out [N,1]
  k_mlp3<8, 0, 1><<<gN, TPB, 0, stream>>>(hB, nullptr,
      w_dec1, b_dec1, w_dec2, b_dec2, w_dec3, b_dec3, (float*)d_out, N);
}
